// MixtureOfExperts_80882824118731
// MI455X (gfx1250) — matrix operand order repaired and checked by reading, not compiled, not run on hardware
//
#include <hip/hip_runtime.h>
#include <hip/hip_bf16.h>
#include <math.h>

#define DIM 512
#define NEXP 8
#define HID 2048
#define NTOK 4096
#define NSLOT (2 * NTOK)

typedef __attribute__((ext_vector_type(16))) __bf16 bf16x16;
typedef __attribute__((ext_vector_type(8)))  float  floatx8;

union Bf16x16U {
    bf16x16 v;
    uint4   q[2];
    unsigned short s[16];
};

union Bf16x8U {
    uint4 q;
    unsigned short s[8];
    __bf16 h[8];
};

// Issue one per-lane async 16B copy global -> LDS (ASYNCcnt-tracked).
// VGLOBAL GVS form: per-lane LDS byte address, per-lane 32-bit global byte
// offset, 64-bit SGPR base.
#define ASYNC_COPY_B128(ldsoff, goff, base)                                   \
    asm volatile("global_load_async_to_lds_b128 %0, %1, %2"                   \
                 :: "v"(ldsoff), "v"(goff), "s"(base) : "memory")

#define WAIT_ASYNC_2() asm volatile("s_wait_asynccnt 0x2" ::: "memory")
#define WAIT_ASYNC_0() asm volatile("s_wait_asynccnt 0x0" ::: "memory")

// ---------------------------------------------------------------------------
// Conversion kernels (run once per call; weights are reused ~16x by the GEMMs
// so paying one f32->bf16 + transpose pass halves streamed bytes and removes
// all conversion VALU from the WMMA inner loops).
// ---------------------------------------------------------------------------

// x [NTOK][DIM] f32 -> xb bf16 (same layout)
__global__ __launch_bounds__(256)
void xcvt_kernel(const float* __restrict__ x, unsigned short* __restrict__ xb)
{
    const size_t idx = ((size_t)blockIdx.x * 256 + threadIdx.x) * 8;
    const float4 a = *(const float4*)(x + idx);
    const float4 b = *(const float4*)(x + idx + 4);
    Bf16x8U u;
    u.h[0] = (__bf16)a.x; u.h[1] = (__bf16)a.y;
    u.h[2] = (__bf16)a.z; u.h[3] = (__bf16)a.w;
    u.h[4] = (__bf16)b.x; u.h[5] = (__bf16)b.y;
    u.h[6] = (__bf16)b.z; u.h[7] = (__bf16)b.w;
    *(uint4*)(xb + idx) = u.q;
}

// src f32 [E][K][N] -> dst bf16 [E][N][K]  (transpose + convert, 64x64 tiles)
__global__ __launch_bounds__(256)
void wcvt_kernel(const float* __restrict__ src, unsigned short* __restrict__ dst,
                 int K, int N)
{
    __shared__ unsigned short T[64 * 65];
    const int n0 = blockIdx.x * 64;
    const int k0 = blockIdx.y * 64;
    const int e  = blockIdx.z;
    const float* s = src + (size_t)e * K * N;
    unsigned short* d = dst + (size_t)e * N * K;
    const int t = threadIdx.x;
#pragma unroll
    for (int i = 0; i < 16; ++i) {
        const int idx = t + i * 256;
        const int kk = idx >> 6;
        const int nn = idx & 63;
        const float v = s[(size_t)(k0 + kk) * N + n0 + nn];
        const __bf16 h = (__bf16)v;
        T[kk * 65 + nn] = *(const unsigned short*)&h;
    }
    __syncthreads();
#pragma unroll
    for (int i = 0; i < 16; ++i) {
        const int idx = t + i * 256;
        const int nn = idx >> 6;
        const int kk = idx & 63;
        d[(size_t)(n0 + nn) * K + k0 + kk] = T[kk * 65 + nn];
    }
}

// ---------------------------------------------------------------------------
// Kernel: routing. One wave (32 lanes) per token.
// ---------------------------------------------------------------------------
__global__ __launch_bounds__(256)
void route_kernel(const float* __restrict__ x, const float* __restrict__ Wr,
                  const float* __restrict__ br,
                  int* __restrict__ cnt, int* __restrict__ topIdx,
                  float* __restrict__ topW, int* __restrict__ topPos)
{
    __shared__ float WrS[DIM * NEXP];
    __shared__ float brS[NEXP];
    const int t = threadIdx.x;
    for (int i = t; i < DIM * NEXP; i += 256) WrS[i] = Wr[i];
    if (t < NEXP) brS[t] = br[t];
    __syncthreads();

    const int wave = t >> 5;
    const int lane = t & 31;
    const int n = blockIdx.x * 8 + wave;

    float part[NEXP];
#pragma unroll
    for (int e = 0; e < NEXP; ++e) part[e] = 0.0f;

    const float* xr = x + (size_t)n * DIM;
#pragma unroll 4
    for (int i = 0; i < DIM / 32; ++i) {
        const int d = lane + i * 32;
        const float xv = xr[d];
#pragma unroll
        for (int e = 0; e < NEXP; ++e) part[e] += xv * WrS[d * NEXP + e];
    }
#pragma unroll
    for (int e = 0; e < NEXP; ++e) {
#pragma unroll
        for (int m = 16; m >= 1; m >>= 1)
            part[e] += __shfl_xor(part[e], m, 32);
        part[e] += brS[e];
    }

    if (lane == 0) {
        int i0 = 0;
#pragma unroll
        for (int e = 1; e < NEXP; ++e) if (part[e] > part[i0]) i0 = e;
        int i1 = (i0 == 0) ? 1 : 0;
#pragma unroll
        for (int e = 0; e < NEXP; ++e)
            if (e != i0 && part[e] > part[i1]) i1 = e;
        const float d = expf(part[i1] - part[i0]);
        const float inv = 1.0f / (1.0f + d);
        const int p0 = atomicAdd(&cnt[i0], 1);
        const int p1 = atomicAdd(&cnt[i1], 1);
        topIdx[2 * n]     = i0;      topIdx[2 * n + 1] = i1;
        topW [2 * n]      = inv;     topW [2 * n + 1]  = d * inv;
        topPos[2 * n]     = p0;      topPos[2 * n + 1] = p1;
    }
}

// ---------------------------------------------------------------------------
// Kernel: scan over 8 expert counts + load / lb_loss outputs.
// ---------------------------------------------------------------------------
__global__ void scan_kernel(const int* __restrict__ cnt, int* __restrict__ off,
                            float* __restrict__ out_tail)
{
    if (threadIdx.x == 0) {
        int o = 0;
        float lb = 0.0f;
        for (int e = 0; e < NEXP; ++e) {
            off[e] = o;
            o += cnt[e];
            const float load = (float)cnt[e] / (float)NSLOT;
            out_tail[1 + e] = load;
            const float d = load - 1.0f / (float)NEXP;
            lb += d * d;
        }
        out_tail[0] = lb;
    }
}

// ---------------------------------------------------------------------------
// Kernel: scatter tokens into compacted per-expert slot ranges.
// ---------------------------------------------------------------------------
__global__ __launch_bounds__(256)
void assign_kernel(const int* __restrict__ topIdx, const float* __restrict__ topW,
                   const int* __restrict__ topPos, const int* __restrict__ off,
                   int* __restrict__ slot_token, float* __restrict__ slot_w)
{
    const int idx = blockIdx.x * 256 + threadIdx.x;
    if (idx >= NSLOT) return;
    const int n = idx >> 1;
    const int e = topIdx[idx];
    const int slot = off[e] + topPos[idx];
    slot_token[slot] = n;
    slot_w[slot] = topW[idx];
}

// ---------------------------------------------------------------------------
// GEMM1: H[slot,:] = GELU( xb[token(slot),:] @ W1[e] + b1[e] )  (bf16 spill)
// Bt1 = W1 pre-transposed bf16 [e][n=HID][k=DIM].
// Block 64x128, 8 waves (2x4), each wave 2x2 16x16x32 bf16 WMMA tiles.
// B tiles double-buffered in LDS via global_load_async_to_lds_b128.
// ---------------------------------------------------------------------------
__global__ __launch_bounds__(256)
void gemm1_kernel(const unsigned short* __restrict__ xb,
                  const unsigned short* __restrict__ Bt1,
                  const float* __restrict__ b1,
                  const int* __restrict__ slot_token,
                  const int* __restrict__ off, const int* __restrict__ cnt,
                  unsigned short* __restrict__ Hbuf)
{
    const int e  = blockIdx.x >> 6;
    const int rt = blockIdx.x & 63;
    const int c0 = cnt[e];
    if (rt * 64 >= c0) return;
    const int base   = off[e];
    const int srow0  = base + rt * 64;
    const int rowEnd = base + c0;
    const int nb0    = blockIdx.y * 128;

    __shared__ __align__(16) unsigned short Bl[2][128 * 32]; // [n][k] bf16

    const int t     = threadIdx.x;
    const int wave  = t >> 5;
    const int lane  = t & 31;
    const int wm    = wave >> 2;
    const int wn    = wave & 3;
    const int lmod  = lane & 15;
    const int lhalf = lane >> 4;

    floatx8 acc[2][2];
#pragma unroll
    for (int a = 0; a < 2; ++a)
#pragma unroll
        for (int b = 0; b < 2; ++b) acc[a][b] = (floatx8)(0.0f);

    // async-copy parameters: tile is 128 rows(n) x 32(k) bf16 = 8 KB,
    // 512 x 16B chunks, 2 per thread. idx -> row = idx>>2, chunk = idx&3.
    const unsigned short* Bbase =
        Bt1 + (size_t)e * HID * DIM + (size_t)nb0 * DIM;
    const unsigned ldsA = (unsigned)(size_t)&Bl[0][0];
    const unsigned ldsB = (unsigned)(size_t)&Bl[1][0];
    const int idx0 = t, idx1 = t + 256;
    const unsigned lo0 = (unsigned)idx0 * 16u;
    const unsigned lo1 = (unsigned)idx1 * 16u;
    const unsigned gstat0 = (unsigned)(idx0 >> 2) * (DIM * 2u) + (unsigned)(idx0 & 3) * 16u;
    const unsigned gstat1 = (unsigned)(idx1 >> 2) * (DIM * 2u) + (unsigned)(idx1 & 3) * 16u;

    int tokRow[2];
#pragma unroll
    for (int mt = 0; mt < 2; ++mt) {
        int sr = srow0 + wm * 32 + mt * 16 + lmod;
        if (sr >= rowEnd) sr = rowEnd - 1;
        tokRow[mt] = slot_token[sr];
    }

    // prologue: fill buffer 0 with k0 = 0
    ASYNC_COPY_B128(ldsA + lo0, gstat0, Bbase);
    ASYNC_COPY_B128(ldsA + lo1, gstat1, Bbase);

    int buf = 0;
    for (int k0 = 0; k0 < DIM; k0 += 32) {
        const bool hasNext = (k0 + 32) < DIM;
        if (hasNext) {
            const unsigned kb = (unsigned)(k0 + 32) * 2u;
            const unsigned lds = (buf == 0) ? ldsB : ldsA;
            ASYNC_COPY_B128(lds + lo0, gstat0 + kb, Bbase);
            ASYNC_COPY_B128(lds + lo1, gstat1 + kb, Bbase);
            WAIT_ASYNC_2();           // current tile (first 2 issued) complete
        } else {
            WAIT_ASYNC_0();
        }
        __syncthreads();

        // A fragments straight from bf16 global (two 16B loads per lane).
        bf16x16 afrag[2];
#pragma unroll
        for (int mt = 0; mt < 2; ++mt) {
            const unsigned short* ar =
                xb + (size_t)tokRow[mt] * DIM + k0 + lhalf * 8;
            Bf16x16U u;
            u.q[0] = *(const uint4*)(ar);       // K khalf+0..7
            u.q[1] = *(const uint4*)(ar + 16);  // K 16+khalf+0..7
            afrag[mt] = u.v;
        }
        // B fragments from LDS: column N=lmod, K-half per lane half.
        bf16x16 bfrag[2];
#pragma unroll
        for (int nt = 0; nt < 2; ++nt) {
            const int c = wn * 32 + nt * 16 + lmod;
            const uint4* p = (const uint4*)&Bl[buf][c * 32 + lhalf * 8];
            Bf16x16U u;
            u.q[0] = p[0];
            u.q[1] = p[2];
            bfrag[nt] = u.v;
        }
#pragma unroll
        for (int mt = 0; mt < 2; ++mt)
#pragma unroll
            for (int nt = 0; nt < 2; ++nt)
                acc[mt][nt] = __builtin_amdgcn_wmma_f32_16x16x32_bf16(
                    false, afrag[mt], false, bfrag[nt],
                    (short)0, acc[mt][nt], false, false);
        __syncthreads();
        buf ^= 1;
    }

    const float* b1e = b1 + e * HID;
#pragma unroll
    for (int mt = 0; mt < 2; ++mt) {
#pragma unroll
        for (int nt = 0; nt < 2; ++nt) {
            const int n = nb0 + wn * 32 + nt * 16 + lmod;
            const float bias = b1e[n];
#pragma unroll
            for (int i = 0; i < 8; ++i) {
                const int sr = srow0 + wm * 32 + mt * 16 + i + lhalf * 8;
                if (sr < rowEnd) {
                    const float v = acc[mt][nt][i] + bias;
                    const float g = 0.5f * v * (1.0f + erff(v * 0.70710678118f));
                    const __bf16 h = (__bf16)g;
                    Hbuf[(size_t)sr * HID + n] = *(const unsigned short*)&h;
                }
            }
        }
    }
}

// ---------------------------------------------------------------------------
// GEMM2: Y[slot,:] = H[slot,:] @ W2[e] + b2[e]; out[token] += w_slot * Y
// Bt2 = W2 pre-transposed bf16 [e][n=DIM][k=HID].
// ---------------------------------------------------------------------------
__global__ __launch_bounds__(256)
void gemm2_kernel(const unsigned short* __restrict__ Hbuf,
                  const unsigned short* __restrict__ Bt2,
                  const float* __restrict__ b2,
                  const int* __restrict__ slot_token, const float* __restrict__ slot_w,
                  const int* __restrict__ off, const int* __restrict__ cnt,
                  float* __restrict__ out)
{
    const int e  = blockIdx.x >> 6;
    const int rt = blockIdx.x & 63;
    const int c0 = cnt[e];
    if (rt * 64 >= c0) return;
    const int base   = off[e];
    const int srow0  = base + rt * 64;
    const int rowEnd = base + c0;
    const int nb0    = blockIdx.y * 128;

    __shared__ __align__(16) unsigned short Bl[2][128 * 32];

    const int t     = threadIdx.x;
    const int wave  = t >> 5;
    const int lane  = t & 31;
    const int wm    = wave >> 2;
    const int wn    = wave & 3;
    const int lmod  = lane & 15;
    const int lhalf = lane >> 4;

    floatx8 acc[2][2];
#pragma unroll
    for (int a = 0; a < 2; ++a)
#pragma unroll
        for (int b = 0; b < 2; ++b) acc[a][b] = (floatx8)(0.0f);

    const unsigned short* Bbase =
        Bt2 + (size_t)e * DIM * HID + (size_t)nb0 * HID;
    const unsigned ldsA = (unsigned)(size_t)&Bl[0][0];
    const unsigned ldsB = (unsigned)(size_t)&Bl[1][0];
    const int idx0 = t, idx1 = t + 256;
    const unsigned lo0 = (unsigned)idx0 * 16u;
    const unsigned lo1 = (unsigned)idx1 * 16u;
    const unsigned gstat0 = (unsigned)(idx0 >> 2) * (HID * 2u) + (unsigned)(idx0 & 3) * 16u;
    const unsigned gstat1 = (unsigned)(idx1 >> 2) * (HID * 2u) + (unsigned)(idx1 & 3) * 16u;

    int srowA[2];
#pragma unroll
    for (int mt = 0; mt < 2; ++mt) {
        int sr = srow0 + wm * 32 + mt * 16 + lmod;
        if (sr >= rowEnd) sr = rowEnd - 1;
        srowA[mt] = sr;
    }

    ASYNC_COPY_B128(ldsA + lo0, gstat0, Bbase);
    ASYNC_COPY_B128(ldsA + lo1, gstat1, Bbase);

    int buf = 0;
    for (int k0 = 0; k0 < HID; k0 += 32) {
        const bool hasNext = (k0 + 32) < HID;
        if (hasNext) {
            const unsigned kb = (unsigned)(k0 + 32) * 2u;
            const unsigned lds = (buf == 0) ? ldsB : ldsA;
            ASYNC_COPY_B128(lds + lo0, gstat0 + kb, Bbase);
            ASYNC_COPY_B128(lds + lo1, gstat1 + kb, Bbase);
            WAIT_ASYNC_2();
        } else {
            WAIT_ASYNC_0();
        }
        __syncthreads();

        bf16x16 afrag[2];
#pragma unroll
        for (int mt = 0; mt < 2; ++mt) {
            const unsigned short* hr =
                Hbuf + (size_t)srowA[mt] * HID + k0 + lhalf * 8;
            Bf16x16U u;
            u.q[0] = *(const uint4*)(hr);
            u.q[1] = *(const uint4*)(hr + 16);
            afrag[mt] = u.v;
        }
        bf16x16 bfrag[2];
#pragma unroll
        for (int nt = 0; nt < 2; ++nt) {
            const int c = wn * 32 + nt * 16 + lmod;
            const uint4* p = (const uint4*)&Bl[buf][c * 32 + lhalf * 8];
            Bf16x16U u;
            u.q[0] = p[0];
            u.q[1] = p[2];
            bfrag[nt] = u.v;
        }
#pragma unroll
        for (int mt = 0; mt < 2; ++mt)
#pragma unroll
            for (int nt = 0; nt < 2; ++nt)
                acc[mt][nt] = __builtin_amdgcn_wmma_f32_16x16x32_bf16(
                    false, afrag[mt], false, bfrag[nt],
                    (short)0, acc[mt][nt], false, false);
        __syncthreads();
        buf ^= 1;
    }

    const float* b2e = b2 + e * DIM;
#pragma unroll
    for (int mt = 0; mt < 2; ++mt) {
#pragma unroll
        for (int nt = 0; nt < 2; ++nt) {
            const int n = nb0 + wn * 32 + nt * 16 + lmod;
            const float bias = b2e[n];
#pragma unroll
            for (int i = 0; i < 8; ++i) {
                const int sr = srow0 + wm * 32 + mt * 16 + i + lhalf * 8;
                if (sr < rowEnd) {
                    const float y = acc[mt][nt][i] + bias;
                    const int tok = slot_token[sr];
                    const float w = slot_w[sr];
                    atomicAdd(&out[(size_t)tok * DIM + n], w * y);
                }
            }
        }
    }
}

// ---------------------------------------------------------------------------
extern "C" void kernel_launch(void* const* d_in, const int* in_sizes, int n_in,
                              void* d_out, int out_size, void* d_ws, size_t ws_size,
                              hipStream_t stream)
{
    const float* x  = (const float*)d_in[0];
    const float* Wr = (const float*)d_in[1];
    const float* br = (const float*)d_in[2];
    const float* W1 = (const float*)d_in[3];
    const float* b1 = (const float*)d_in[4];
    const float* W2 = (const float*)d_in[5];
    const float* b2 = (const float*)d_in[6];
    float* out = (float*)d_out;

    char* w = (char*)d_ws;
    int*   cnt        = (int*)w;             w += 256;
    int*   off        = (int*)w;             w += 256;
    int*   topIdx     = (int*)w;             w += NSLOT * 4;
    float* topW       = (float*)w;           w += NSLOT * 4;
    int*   topPos     = (int*)w;             w += NSLOT * 4;
    int*   slot_token = (int*)w;             w += NSLOT * 4;
    float* slot_w     = (float*)w;           w += NSLOT * 4;
    unsigned short* xb   = (unsigned short*)w;  w += (size_t)NTOK * DIM * 2;
    unsigned short* Bt1  = (unsigned short*)w;  w += (size_t)NEXP * DIM * HID * 2;
    unsigned short* Bt2  = (unsigned short*)w;  w += (size_t)NEXP * HID * DIM * 2;
    unsigned short* Hbuf = (unsigned short*)w;  // NSLOT * HID bf16 = 33.5 MB

    hipMemsetAsync(cnt, 0, 256, stream);
    hipMemsetAsync(d_out, 0, (size_t)out_size * sizeof(float), stream);

    // one-time (per call) precision / layout conversion
    xcvt_kernel<<<(NTOK * DIM) / (256 * 8), 256, 0, stream>>>(x, xb);
    wcvt_kernel<<<dim3(HID / 64, DIM / 64, NEXP), 256, 0, stream>>>(W1, Bt1, DIM, HID);
    wcvt_kernel<<<dim3(DIM / 64, HID / 64, NEXP), 256, 0, stream>>>(W2, Bt2, HID, DIM);

    route_kernel<<<NTOK / 8, 256, 0, stream>>>(x, Wr, br, cnt, topIdx, topW, topPos);
    scan_kernel<<<1, 32, 0, stream>>>(cnt, off, out + (size_t)NTOK * DIM);
    assign_kernel<<<NSLOT / 256, 256, 0, stream>>>(topIdx, topW, topPos, off,
                                                   slot_token, slot_w);
    gemm1_kernel<<<dim3(NEXP * 64, HID / 128), 256, 0, stream>>>(
        xb, Bt1, b1, slot_token, off, cnt, Hbuf);
    gemm2_kernel<<<dim3(NEXP * 64, DIM / 128), 256, 0, stream>>>(
        Hbuf, Bt2, b2, slot_token, slot_w, off, cnt, out);
}
